// AoAGNN_12867722019083
// MI455X (gfx1250) — compile-verified, matrix operand order checked by reading, not run on hardware
//
#include <hip/hip_runtime.h>
#include <hip/hip_bf16.h>
#include <math.h>

// ---------------- problem constants ----------------
#define N_NODES 50000
#define N_EDGES 500000
#define IN_DIM  256
#define HID     128
#define HEADS   2
#define N_GRAPH 64
#define SLOPE   0.2f
#define ETOT    (N_EDGES + N_NODES)       /* self loops appended */
#define NPAD    50016                     /* (N/16 + 1)*16 */

typedef __attribute__((ext_vector_type(16))) _Float16 v16h;
typedef __attribute__((ext_vector_type(8)))  float    v8f;

// ---------------- small device helpers ----------------
__device__ inline float lrelu(float x) { return x > 0.f ? x : SLOPE * x; }

__device__ inline float waveReduceSum(float v) {
#pragma unroll
  for (int off = 16; off > 0; off >>= 1) v += __shfl_xor(v, off, 32);
  return v;
}

__device__ inline void atomicMaxFloat(float* addr, float val) {
  if (!(val < 0.0f)) atomicMax((int*)addr, __float_as_int(val));
  else               atomicMin((unsigned int*)addr, __float_as_uint(val));
}

// ---------------- utility kernels ----------------
__global__ void fill_f32(float* __restrict__ p, float v, int n) {
  int i = blockIdx.x * blockDim.x + threadIdx.x;
  if (i < n) p[i] = v;
}

// convert fp32 -> f16, zero-padding rows >= rows (dst has rowsPad rows)
__global__ void cvt_f16_pad(_Float16* __restrict__ dst, const float* __restrict__ src,
                            int rows, int rowsPad, int cols) {
  int i = blockIdx.x * blockDim.x + threadIdx.x;
  int total = rowsPad * cols;
  if (i >= total) return;
  int r = i / cols;
  dst[i] = (r < rows) ? (_Float16)src[i] : (_Float16)0.0f;
}

// ---------------- WMMA GEMM: C[Np,M] = act(A16[Np,K] @ W16[M,K]^T + bias) ----------------
// one wave per 16-row tile; loops over M/16 col tiles, K in slabs of 32.
// WF/WH compile-time: which outputs to write (f32 / f16) -> branch-free epilogue.
template <int K, int M, int ACT, bool WF, bool WH>
__global__ void __launch_bounds__(128)
gemm_wmma_f16(const _Float16* __restrict__ A, const _Float16* __restrict__ W,
              const float* __restrict__ bias,
              float* __restrict__ outF, _Float16* __restrict__ outH, int ntiles) {
  int wave = blockIdx.x * (blockDim.x >> 5) + (threadIdx.x >> 5);
  if (wave >= ntiles) return;                 // wave-uniform: EXEC stays all-ones
  int lane = threadIdx.x & 31;
  int r    = lane & 15;                       // row within 16-tile (A) / col (B)
  int koff = (lane >> 4) << 4;                // K half-slice select per lane group
  const _Float16* arow = A + (size_t)(wave * 16 + r) * K + koff;

  // preload all A fragments for this 16-row tile (loop-invariant over col tiles)
  v16h afr[K / 32];
#pragma unroll
  for (int ks = 0; ks < K / 32; ++ks) afr[ks] = *(const v16h*)(arow + ks * 32);

#pragma unroll 1
  for (int nt = 0; nt < M / 16; ++nt) {
    v8f acc = {};
    const _Float16* brow = W + (size_t)(nt * 16 + r) * K + koff;
#pragma unroll
    for (int ks = 0; ks < K / 32; ++ks) {
      v16h b = *(const v16h*)(brow + ks * 32);
      acc = __builtin_amdgcn_wmma_f32_16x16x32_f16(
          false, afr[ks], false, b, (short)0, acc, false, false);
    }
    int col   = nt * 16 + r;
    float bv  = bias[col];
    int mbase = wave * 16 + ((lane >> 4) << 3);   // lanes 16-31 hold M rows 8..15
#pragma unroll
    for (int v = 0; v < 8; ++v) {
      float val = acc[v] + bv;
      if (ACT) val = val > 0.f ? val : 0.f;
      size_t idx = (size_t)(mbase + v) * M + col;
      if (WF) outF[idx] = val;
      if (WH) outH[idx] = (_Float16)val;
    }
  }
}

// ---------------- GATv2 edge kernels ----------------
// one wave per (edge, head): e = att . leaky_relu(xl[src]+xr[dst]); segment max
__global__ void __launch_bounds__(128)
edge_logits(const float* __restrict__ xl, const float* __restrict__ xr,
            const int* __restrict__ ei, const float* __restrict__ att,
            float* __restrict__ elog, float* __restrict__ m) {
  int wid = blockIdx.x * 4 + (threadIdx.x >> 5);
  if (wid >= ETOT * HEADS) return;
  int lane = threadIdx.x & 31;
  int e = wid >> 1, h = wid & 1;
  int s, d;
  if (e < N_EDGES) { s = ei[e]; d = ei[N_EDGES + e]; } else { s = d = e - N_EDGES; }
  const float* pl = xl + (size_t)s * (HEADS * HID) + h * HID;
  const float* pr = xr + (size_t)d * (HEADS * HID) + h * HID;
  const float* pa = att + h * HID;
  int c = lane * 4;
  float4 vl = *(const float4*)(pl + c);
  float4 vr = *(const float4*)(pr + c);
  float4 va = *(const float4*)(pa + c);
  float acc = lrelu(vl.x + vr.x) * va.x + lrelu(vl.y + vr.y) * va.y +
              lrelu(vl.z + vr.z) * va.z + lrelu(vl.w + vr.w) * va.w;
  acc = waveReduceSum(acc);
  if (lane == 0) {
    elog[wid] = acc;
    atomicMaxFloat(&m[d * HEADS + h], acc);
  }
}

// fused: a = exp(e - m[dst]); ssum[dst,h] += a; aggr[dst,h,:] += a * xl[src,h,:]
// (normalization deferred to head_mean: sum(a_i/S * x_i) == (sum a_i x_i)/S)
__global__ void __launch_bounds__(128)
edge_softmax_scatter(const float* __restrict__ xl, const float* __restrict__ elog,
                     const float* __restrict__ m, const int* __restrict__ ei,
                     float* __restrict__ ssum, float* __restrict__ aggr) {
  int wid = blockIdx.x * 4 + (threadIdx.x >> 5);
  if (wid >= ETOT * HEADS) return;
  int lane = threadIdx.x & 31;
  int e = wid >> 1, h = wid & 1;
  int s, d;
  if (e < N_EDGES) { s = ei[e]; d = ei[N_EDGES + e]; } else { s = d = e - N_EDGES; }
  float a = __expf(elog[wid] - m[d * HEADS + h]);
  if (lane == 0) atomicAdd(&ssum[d * HEADS + h], a);
  const float* pl = xl + (size_t)s * (HEADS * HID) + h * HID;
  float* po = aggr + (size_t)d * (HEADS * HID) + h * HID;
  int c = lane * 4;
  float4 v = *(const float4*)(pl + c);
  atomicAdd(po + c + 0, a * v.x);
  atomicAdd(po + c + 1, a * v.y);
  atomicAdd(po + c + 2, a * v.z);
  atomicAdd(po + c + 3, a * v.w);
}

// h = relu(mean over heads of aggr/ssum + bias); emit f16 (always) and optional f32
template <bool WF>
__global__ void head_mean(const float* __restrict__ aggr, const float* __restrict__ ssum,
                          const float* __restrict__ bias,
                          float* __restrict__ hF, _Float16* __restrict__ hH) {
  int i = blockIdx.x * blockDim.x + threadIdx.x;
  if (i >= NPAD * HID) return;
  int c = i & (HID - 1);
  int n = i >> 7;
  float s0 = ssum[n * HEADS + 0] + 1e-16f;
  float s1 = ssum[n * HEADS + 1] + 1e-16f;
  float val = 0.5f * (aggr[(size_t)n * (HEADS * HID) + c] / s0 +
                      aggr[(size_t)n * (HEADS * HID) + HID + c] / s1) + bias[c];
  val = val > 0.f ? val : 0.f;
  if (WF) hF[i] = val;
  hH[i] = (_Float16)val;
}

// ---------------- pooling kernels ----------------
// gate[n] = gate1[n,:] . w2 + b2 ; segment max over batch (one wave / node)
__global__ void __launch_bounds__(128)
gate_score(const float* __restrict__ g1, const float* __restrict__ w2,
           const float* __restrict__ b2, const int* __restrict__ batch,
           float* __restrict__ gate, float* __restrict__ gm) {
  int wid = blockIdx.x * 4 + (threadIdx.x >> 5);
  if (wid >= N_NODES) return;
  int lane = threadIdx.x & 31;
  int c = lane * 4;
  float4 v = *(const float4*)(g1 + (size_t)wid * HID + c);
  float4 w = *(const float4*)(w2 + c);
  float acc = v.x * w.x + v.y * w.y + v.z * w.z + v.w * w.w;
  acc = waveReduceSum(acc);
  if (lane == 0) {
    acc += b2[0];
    gate[wid] = acc;
    atomicMaxFloat(&gm[batch[wid]], acc);
  }
}

// fused: a = exp(gate - gm[b]); gs[b] += a; gpool[b,:] += a * h[n,:]
__global__ void __launch_bounds__(128)
pool_softmax_scatter(const float* __restrict__ gate, const float* __restrict__ gm,
                     const int* __restrict__ batch, const float* __restrict__ hF,
                     float* __restrict__ gs, float* __restrict__ gpool) {
  int wid = blockIdx.x * 4 + (threadIdx.x >> 5);
  if (wid >= N_NODES) return;
  int lane = threadIdx.x & 31;
  int b = batch[wid];
  float a = __expf(gate[wid] - gm[b]);
  if (lane == 0) atomicAdd(&gs[b], a);
  int c = lane * 4;
  float4 v = *(const float4*)(hF + (size_t)wid * HID + c);
  float* po = gpool + (size_t)b * HID + c;
  atomicAdd(po + 0, a * v.x);
  atomicAdd(po + 1, a * v.y);
  atomicAdd(po + 2, a * v.z);
  atomicAdd(po + 3, a * v.w);
}

// g16[b,c] = (gpool[b,c] / (gs[b]+eps)) as f16
__global__ void pool_final(const float* __restrict__ gpool, const float* __restrict__ gs,
                           _Float16* __restrict__ g16) {
  int i = blockIdx.x * blockDim.x + threadIdx.x;
  if (i >= N_GRAPH * HID) return;
  int b = i >> 7;
  g16[i] = (_Float16)(gpool[i] / (gs[b] + 1e-16f));
}

// final: out = normalize(h1 @ head_w2^T + b2)  (one wave / graph, 2 outputs)
__global__ void __launch_bounds__(128)
head_final(const float* __restrict__ h1, const float* __restrict__ w2,
           const float* __restrict__ b2, float* __restrict__ out) {
  int wid = blockIdx.x * 4 + (threadIdx.x >> 5);
  if (wid >= N_GRAPH) return;
  int lane = threadIdx.x & 31;
  int c = lane * 4;
  float4 v  = *(const float4*)(h1 + (size_t)wid * HID + c);
  float4 wa = *(const float4*)(w2 + c);
  float4 wb = *(const float4*)(w2 + HID + c);
  float a0 = v.x * wa.x + v.y * wa.y + v.z * wa.z + v.w * wa.w;
  float a1 = v.x * wb.x + v.y * wb.y + v.z * wb.z + v.w * wb.w;
  a0 = waveReduceSum(a0);
  a1 = waveReduceSum(a1);
  if (lane == 0) {
    float o0 = a0 + b2[0], o1 = a1 + b2[1];
    float nrm = fmaxf(sqrtf(o0 * o0 + o1 * o1), 1e-12f);
    out[wid * 2 + 0] = o0 / nrm;
    out[wid * 2 + 1] = o1 / nrm;
  }
}

// ---------------- launcher ----------------
static inline size_t alignUp(size_t x) { return (x + 255) & ~(size_t)255; }

extern "C" void kernel_launch(void* const* d_in, const int* in_sizes, int n_in,
                              void* d_out, int out_size, void* d_ws, size_t ws_size,
                              hipStream_t stream) {
  (void)in_sizes; (void)n_in; (void)out_size; (void)ws_size;
  const float* x        = (const float*)d_in[0];
  const int*   ei       = (const int*)d_in[1];
  const int*   batch    = (const int*)d_in[2];
  const float* enc_w1   = (const float*)d_in[3];
  const float* enc_b1   = (const float*)d_in[4];
  const float* enc_w2   = (const float*)d_in[5];
  const float* enc_b2   = (const float*)d_in[6];
  const float* c1_wl    = (const float*)d_in[7];
  const float* c1_bl    = (const float*)d_in[8];
  const float* c1_wr    = (const float*)d_in[9];
  const float* c1_br    = (const float*)d_in[10];
  const float* c1_att   = (const float*)d_in[11];
  const float* c1_bias  = (const float*)d_in[12];
  const float* c2_wl    = (const float*)d_in[13];
  const float* c2_bl    = (const float*)d_in[14];
  const float* c2_wr    = (const float*)d_in[15];
  const float* c2_br    = (const float*)d_in[16];
  const float* c2_att   = (const float*)d_in[17];
  const float* c2_bias  = (const float*)d_in[18];
  const float* gate_w1  = (const float*)d_in[19];
  const float* gate_b1  = (const float*)d_in[20];
  const float* gate_w2  = (const float*)d_in[21];
  const float* gate_b2  = (const float*)d_in[22];
  const float* head_w1  = (const float*)d_in[23];
  const float* head_b1  = (const float*)d_in[24];
  const float* head_w2  = (const float*)d_in[25];
  const float* head_b2  = (const float*)d_in[26];
  float* out = (float*)d_out;

  // ---- carve workspace ----
  char* p = (char*)d_ws;
  auto carve = [&](size_t bytes) { char* r = p; p += alignUp(bytes); return (void*)r; };
  _Float16* x16    = (_Float16*)carve((size_t)NPAD * IN_DIM * 2);
  _Float16* wE1    = (_Float16*)carve(HID * IN_DIM * 2);
  _Float16* wE2    = (_Float16*)carve(HID * HID * 2);
  _Float16* wL1    = (_Float16*)carve(HEADS * HID * HID * 2);
  _Float16* wR1    = (_Float16*)carve(HEADS * HID * HID * 2);
  _Float16* wL2    = (_Float16*)carve(HEADS * HID * HID * 2);
  _Float16* wR2    = (_Float16*)carve(HEADS * HID * HID * 2);
  _Float16* wG1    = (_Float16*)carve(HID * HID * 2);
  _Float16* wH1    = (_Float16*)carve(HID * HID * 2);
  _Float16* h16a   = (_Float16*)carve((size_t)NPAD * HID * 2);
  _Float16* h16b   = (_Float16*)carve((size_t)NPAD * HID * 2);
  float* xl        = (float*)carve((size_t)NPAD * HEADS * HID * 4);
  float* xr        = (float*)carve((size_t)NPAD * HEADS * HID * 4);
  float* aggr      = (float*)carve((size_t)NPAD * HEADS * HID * 4);
  float* elog      = (float*)carve((size_t)ETOT * HEADS * 4);
  float* segm      = (float*)carve((size_t)NPAD * HEADS * 4);
  float* segs      = (float*)carve((size_t)NPAD * HEADS * 4);
  float* hF        = (float*)carve((size_t)NPAD * HID * 4);
  float* gate      = (float*)carve((size_t)N_NODES * 4);
  float* gm        = (float*)carve(N_GRAPH * 4);
  float* gs        = (float*)carve(N_GRAPH * 4);
  float* gpool     = (float*)carve(N_GRAPH * HID * 4);
  _Float16* g16    = (_Float16*)carve(N_GRAPH * HID * 2);
  float* head1F    = (float*)carve(N_GRAPH * HID * 4);
  float* gate1F    = xl;  // reuse: free after conv2 edge phase

  const int ntiles = NPAD / 16;             // 3126 row tiles
  const int gemmBlocks = (ntiles + 3) / 4;
  const int nEH = ETOT * HEADS;
  const int waveBlocksE = (nEH + 3) / 4;
  const int waveBlocksN = (N_NODES + 3) / 4;

  // ---- precision conversion ----
  {
    int tot = NPAD * IN_DIM;
    cvt_f16_pad<<<(tot + 255) / 256, 256, 0, stream>>>(x16, x, N_NODES, NPAD, IN_DIM);
    cvt_f16_pad<<<(HID * IN_DIM + 255) / 256, 256, 0, stream>>>(wE1, enc_w1, HID, HID, IN_DIM);
    cvt_f16_pad<<<(HID * HID + 255) / 256, 256, 0, stream>>>(wE2, enc_w2, HID, HID, HID);
    int wsz = HEADS * HID * HID;
    cvt_f16_pad<<<(wsz + 255) / 256, 256, 0, stream>>>(wL1, c1_wl, HEADS * HID, HEADS * HID, HID);
    cvt_f16_pad<<<(wsz + 255) / 256, 256, 0, stream>>>(wR1, c1_wr, HEADS * HID, HEADS * HID, HID);
    cvt_f16_pad<<<(wsz + 255) / 256, 256, 0, stream>>>(wL2, c2_wl, HEADS * HID, HEADS * HID, HID);
    cvt_f16_pad<<<(wsz + 255) / 256, 256, 0, stream>>>(wR2, c2_wr, HEADS * HID, HEADS * HID, HID);
    cvt_f16_pad<<<(HID * HID + 255) / 256, 256, 0, stream>>>(wG1, gate_w1, HID, HID, HID);
    cvt_f16_pad<<<(HID * HID + 255) / 256, 256, 0, stream>>>(wH1, head_w1, HID, HID, HID);
  }

  // ---- encoder MLP (WMMA) ----
  gemm_wmma_f16<IN_DIM, HID, 1, false, true>
      <<<gemmBlocks, 128, 0, stream>>>(x16, wE1, enc_b1, nullptr, h16a, ntiles);
  gemm_wmma_f16<HID, HID, 1, false, true>
      <<<gemmBlocks, 128, 0, stream>>>(h16a, wE2, enc_b2, nullptr, h16b, ntiles);

  // ---- GATv2 conv 1 (input h16b -> output h16a) ----
  gemm_wmma_f16<HID, HEADS * HID, 0, true, false>
      <<<gemmBlocks, 128, 0, stream>>>(h16b, wL1, c1_bl, xl, nullptr, ntiles);
  gemm_wmma_f16<HID, HEADS * HID, 0, true, false>
      <<<gemmBlocks, 128, 0, stream>>>(h16b, wR1, c1_br, xr, nullptr, ntiles);
  fill_f32<<<(NPAD * HEADS + 255) / 256, 256, 0, stream>>>(segm, -INFINITY, NPAD * HEADS);
  fill_f32<<<(NPAD * HEADS + 255) / 256, 256, 0, stream>>>(segs, 0.f, NPAD * HEADS);
  fill_f32<<<(NPAD * HEADS * HID + 255) / 256, 256, 0, stream>>>(aggr, 0.f, NPAD * HEADS * HID);
  edge_logits<<<waveBlocksE, 128, 0, stream>>>(xl, xr, ei, c1_att, elog, segm);
  edge_softmax_scatter<<<waveBlocksE, 128, 0, stream>>>(xl, elog, segm, ei, segs, aggr);
  head_mean<false><<<(NPAD * HID + 255) / 256, 256, 0, stream>>>(aggr, segs, c1_bias, nullptr, h16a);

  // ---- GATv2 conv 2 (input h16a -> output h16b + hF) ----
  gemm_wmma_f16<HID, HEADS * HID, 0, true, false>
      <<<gemmBlocks, 128, 0, stream>>>(h16a, wL2, c2_bl, xl, nullptr, ntiles);
  gemm_wmma_f16<HID, HEADS * HID, 0, true, false>
      <<<gemmBlocks, 128, 0, stream>>>(h16a, wR2, c2_br, xr, nullptr, ntiles);
  fill_f32<<<(NPAD * HEADS + 255) / 256, 256, 0, stream>>>(segm, -INFINITY, NPAD * HEADS);
  fill_f32<<<(NPAD * HEADS + 255) / 256, 256, 0, stream>>>(segs, 0.f, NPAD * HEADS);
  fill_f32<<<(NPAD * HEADS * HID + 255) / 256, 256, 0, stream>>>(aggr, 0.f, NPAD * HEADS * HID);
  edge_logits<<<waveBlocksE, 128, 0, stream>>>(xl, xr, ei, c2_att, elog, segm);
  edge_softmax_scatter<<<waveBlocksE, 128, 0, stream>>>(xl, elog, segm, ei, segs, aggr);
  head_mean<true><<<(NPAD * HID + 255) / 256, 256, 0, stream>>>(aggr, segs, c2_bias, hF, h16b);

  // ---- attentional pooling ----
  gemm_wmma_f16<HID, HID, 1, true, false>
      <<<gemmBlocks, 128, 0, stream>>>(h16b, wG1, gate_b1, gate1F, nullptr, ntiles);
  fill_f32<<<1, 256, 0, stream>>>(gm, -INFINITY, N_GRAPH);
  fill_f32<<<1, 256, 0, stream>>>(gs, 0.f, N_GRAPH);
  fill_f32<<<(N_GRAPH * HID + 255) / 256, 256, 0, stream>>>(gpool, 0.f, N_GRAPH * HID);
  gate_score<<<waveBlocksN, 128, 0, stream>>>(gate1F, gate_w2, gate_b2, batch, gate, gm);
  pool_softmax_scatter<<<waveBlocksN, 128, 0, stream>>>(gate, gm, batch, hF, gs, gpool);
  pool_final<<<(N_GRAPH * HID + 255) / 256, 256, 0, stream>>>(gpool, gs, g16);

  // ---- head MLP + normalize ----
  gemm_wmma_f16<HID, HID, 1, true, false>
      <<<1, 128, 0, stream>>>(g16, wH1, head_b1, head1F, nullptr, N_GRAPH / 16);
  head_final<<<(N_GRAPH + 3) / 4, 128, 0, stream>>>(head1F, head_w2, head_b2, out);
}
